// Grok1Attention_2542620639596
// MI455X (gfx1250) — compile-verified, matrix operand order checked by reading, not run on hardware
//
#include <hip/hip_runtime.h>

typedef __bf16 bf16_t;
typedef __attribute__((ext_vector_type(16))) __bf16 v16bf;
typedef __attribute__((ext_vector_type(8)))  float  v8f;

#define SEQ 2048
#define HIDDEN 4096
#define NHEADS 32
#define NKV 8
#define HDIM 128

__device__ __forceinline__ bf16_t f2bf(float f) {
  union { float f; unsigned u; } v; v.f = f;
  unsigned r = v.u + 0x7FFFu + ((v.u >> 16) & 1u);   // round-to-nearest-even
  unsigned short h = (unsigned short)(r >> 16);
  return __builtin_bit_cast(bf16_t, h);
}

// Fragment-ready K permutation: within each 32-element group, reorder
// [0..7,8..15,16..23,24..31] -> [0..7,16..23,8..15,24..31] (swap middle blocks).
// With this layout a lane's v16bf WMMA operand is one contiguous 32B load:
// lanes 0-15 read group offset 0, lanes 16-31 read group offset 16.
__device__ __forceinline__ int pcol(int c) {
  int b = (c >> 3) & 3;
  b = ((b & 1) << 1) | (b >> 1);
  return (c & ~31) | (b << 3) | (c & 7);
}

// ---------------- fp32 -> bf16, permuted-K layout (rows of length HIDDEN) ----------------
__global__ void cvt_bf16_kernel(const float* __restrict__ in, bf16_t* __restrict__ out, int n) {
  int i = blockIdx.x * blockDim.x + threadIdx.x;
  if (i >= n) return;
  int k = i & (HIDDEN - 1);
  out[(size_t)(i - k) + pcol(k)] = f2bf(in[i]);
}

// ---------------- f32 [K][N] -> bf16 [N][K-permuted] ----------------
__global__ __launch_bounds__(256) void transpose_cvt_kernel(const float* __restrict__ in,
                                                            bf16_t* __restrict__ out,
                                                            int K, int N) {
  __shared__ float tile[32][33];
  int n0 = blockIdx.x * 32, k0 = blockIdx.y * 32;
  int tx = threadIdx.x, ty = threadIdx.y; // 32 x 8
#pragma unroll
  for (int i = 0; i < 4; ++i)
    tile[ty + i * 8][tx] = in[(size_t)(k0 + ty + i * 8) * N + n0 + tx];
  __syncthreads();
#pragma unroll
  for (int i = 0; i < 4; ++i)
    out[(size_t)(n0 + ty + i * 8) * K + pcol(k0 + tx)] = f2bf(tile[tx][ty + i * 8]);
}

// ---------------- bf16 WMMA GEMM: C[M][N] = A[M][Kp] * BT[N][Kp]^T ----------------
// A and BT stored with permuted K. Block tile 128x128, wave tile 32x64,
// both operands staged in LDS (straight 32B copies), 8 WMMA per K-step per wave.
template <bool WRITE_BF16, bool PERM_COL>
__global__ __launch_bounds__(256) void gemm_bf16_kernel(const bf16_t* __restrict__ A,
                                                        const bf16_t* __restrict__ BT,
                                                        void* __restrict__ Cv,
                                                        int M, int N, int K) {
  __shared__ alignas(64) bf16_t As[128 * 32];
  __shared__ alignas(64) bf16_t Bs[128 * 32];
  const int tid = threadIdx.x;
  const int lane = tid & 31;
  const int w = tid >> 5;
  const int n0 = blockIdx.x * 128;
  const int m0 = blockIdx.y * 128;
  const int wr = (w >> 1) * 32;       // wave row offset in block tile
  const int wc = (w & 1) * 64;        // wave col offset in block tile
  const int nl = lane & 15;
  const int ko = (lane & 16) ? 16 : 0;  // permuted-group fragment offset
  const int sr = tid >> 1;            // stage row 0..127
  const int sc = (tid & 1) * 16;      // stage chunk (32B)

  v8f acc[2][4] = {};

  for (int k0 = 0; k0 < K; k0 += 32) {
    *(v16bf*)&As[sr * 32 + sc] = *(const v16bf*)&A[(size_t)(m0 + sr) * K + k0 + sc];
    *(v16bf*)&Bs[sr * 32 + sc] = *(const v16bf*)&BT[(size_t)(n0 + sr) * K + k0 + sc];
    __syncthreads();
    v16bf a0 = *(const v16bf*)&As[(wr + nl) * 32 + ko];
    v16bf a1 = *(const v16bf*)&As[(wr + 16 + nl) * 32 + ko];
#pragma unroll
    for (int nt = 0; nt < 4; ++nt) {
      v16bf b = *(const v16bf*)&Bs[(wc + nt * 16 + nl) * 32 + ko];
      acc[0][nt] = __builtin_amdgcn_wmma_f32_16x16x32_bf16(false, a0, false, b,
                                                           (short)0, acc[0][nt], false, false);
      acc[1][nt] = __builtin_amdgcn_wmma_f32_16x16x32_bf16(false, a1, false, b,
                                                           (short)0, acc[1][nt], false, false);
    }
    __syncthreads();
  }

  const int rbase = (lane & 16) ? 8 : 0;
#pragma unroll
  for (int mi = 0; mi < 2; ++mi)
#pragma unroll
    for (int nt = 0; nt < 4; ++nt) {
      int col = n0 + wc + nt * 16 + nl;
      int colw = PERM_COL ? pcol(col) : col;
#pragma unroll
      for (int r = 0; r < 8; ++r) {
        int rr = m0 + wr + mi * 16 + rbase + r;
        if (WRITE_BF16) ((bf16_t*)Cv)[(size_t)rr * N + colw] = f2bf(acc[mi][nt][r]);
        else            ((float*)Cv)[(size_t)rr * N + colw] = acc[mi][nt][r];
      }
    }
}

// ---------------- in-place NeoX RoPE on bf16 [T][nheads*128], permuted-d layout ----------------
__global__ void rope_kernel(const int* __restrict__ positions, bf16_t* __restrict__ x, int nheads) {
  int idx = blockIdx.x * blockDim.x + threadIdx.x;
  int total = SEQ * nheads * 64;
  if (idx >= total) return;
  int i = idx & 63;
  int h = (idx >> 6) % nheads;
  int t = idx / (nheads * 64);
  float pos = (float)positions[t];
  float inv = __powf(10000.0f, -(float)(2 * i) / 128.0f);
  float fr = pos * inv;
  float s, c;
  __sincosf(fr, &s, &c);
  bf16_t* rowp = x + (size_t)t * nheads * HDIM + h * HDIM;
  int c1 = pcol(i), c2 = pcol(i + 64);
  float x1 = (float)rowp[c1];
  float x2 = (float)rowp[c2];
  rowp[c1] = f2bf(x1 * c - x2 * s);
  rowp[c2] = f2bf(x2 * c + x1 * s);
}

// ---------------- V transpose: [t][hk*128+d] -> [hk][d][t-permuted] ----------------
__global__ void vtrans_kernel(const bf16_t* __restrict__ vb, bf16_t* __restrict__ vT) {
  int idx = blockIdx.x * blockDim.x + threadIdx.x;
  if (idx >= NKV * HDIM * SEQ) return;
  int t = idx & (SEQ - 1);
  int d = (idx >> 11) & (HDIM - 1);
  int hk = idx >> 18;
  vT[((size_t)hk * HDIM + d) * SEQ + pcol(t)] = vb[(size_t)t * (NKV * HDIM) + hk * HDIM + d];
}

// ---------------- flash attention, 1 wave = 16 query rows, 32 keys/step ----------------
__global__ __launch_bounds__(256) void attn_kernel(const bf16_t* __restrict__ Q,   // [T][4096], perm-d
                                                   const bf16_t* __restrict__ Kc,  // [T][1024], perm-d
                                                   const bf16_t* __restrict__ VT,  // [8][128][T], perm-t
                                                   bf16_t* __restrict__ O) {       // [T][4096], perm-col
  __shared__ alignas(64) bf16_t Ps[8][16 * 32];  // per-wave P bounce (C-layout -> A-layout)
  const int tid = threadIdx.x, lane = tid & 31, w = tid >> 5;
  const int h = blockIdx.y;
  const int hkv = h >> 2;            // GQA: 32 Q heads -> 8 KV heads
  const int m0 = blockIdx.x * 128 + w * 16;
  const int nl = lane & 15;
  const int ko = (lane & 16) ? 16 : 0;
  const int rbase = (lane & 16) ? 8 : 0;
  const float scale = 0.08838834764831845f;  // 1/sqrt(128)

  // Q fragments: 16 rows x 128, four K=32 chunks, resident in VGPRs
  v16bf qa[4];
  {
    const bf16_t* qrow = Q + (size_t)(m0 + nl) * (NHEADS * HDIM) + h * HDIM;
#pragma unroll
    for (int c = 0; c < 4; ++c) qa[c] = *(const v16bf*)(qrow + c * 32 + ko);
  }

  v8f o[8] = {};
  float mrow[8], lrow[8];
#pragma unroll
  for (int r = 0; r < 8; ++r) { mrow[r] = -1e30f; lrow[r] = 0.0f; }

  const int kend = m0 + 16;  // causal: keys 0 .. m0+15
  for (int t0 = 0; t0 < kend; t0 += 32) {
    // S = Q K^T : two 16x16 score tiles (32 keys)
    v8f s[2] = {};
#pragma unroll
    for (int st = 0; st < 2; ++st) {
      const bf16_t* krow = Kc + (size_t)(t0 + st * 16 + nl) * (NKV * HDIM) + hkv * HDIM;
#pragma unroll
      for (int c = 0; c < 4; ++c) {
        v16bf kb = *(const v16bf*)(krow + c * 32 + ko);
        s[st] = __builtin_amdgcn_wmma_f32_16x16x32_bf16(false, qa[c], false, kb,
                                                        (short)0, s[st], false, false);
      }
    }

    // causal mask + online softmax (row reductions within 16-lane groups)
#pragma unroll
    for (int r = 0; r < 8; ++r) {
      int qi = m0 + rbase + r;
#pragma unroll
      for (int st = 0; st < 2; ++st) {
        int kk = t0 + st * 16 + nl;
        float v = s[st][r] * scale;
        s[st][r] = (kk <= qi) ? v : -1e30f;
      }
      float mx = fmaxf(s[0][r], s[1][r]);
#pragma unroll
      for (int off = 1; off < 16; off <<= 1) mx = fmaxf(mx, __shfl_xor(mx, off, 16));
      float mnew = fmaxf(mrow[r], mx);
      float alpha = __expf(mrow[r] - mnew);
      mrow[r] = mnew;
      float p0 = __expf(s[0][r] - mnew);
      float p1 = __expf(s[1][r] - mnew);
      float rs = p0 + p1;
#pragma unroll
      for (int off = 1; off < 16; off <<= 1) rs += __shfl_xor(rs, off, 16);
      lrow[r] = lrow[r] * alpha + rs;
      s[0][r] = p0; s[1][r] = p1;
#pragma unroll
      for (int nt = 0; nt < 8; ++nt) o[nt][r] *= alpha;
    }

    // C-layout P -> LDS (permuted key positions) -> contiguous A-layout fragment
    bf16_t* ps = &Ps[w][0];
#pragma unroll
    for (int st = 0; st < 2; ++st)
#pragma unroll
      for (int r = 0; r < 8; ++r)
        ps[(rbase + r) * 32 + pcol(st * 16 + nl)] = f2bf(s[st][r]);
    asm volatile("s_wait_dscnt 0" ::: "memory");
    v16bf pa = *(const v16bf*)(ps + nl * 32 + ko);

    // O += P * V   (VT perm-t gives one contiguous 32B fragment per d-column)
#pragma unroll
    for (int nt = 0; nt < 8; ++nt) {
      const bf16_t* vrow = VT + ((size_t)hkv * HDIM + nt * 16 + nl) * SEQ + t0;
      v16bf vb = *(const v16bf*)(vrow + ko);
      o[nt] = __builtin_amdgcn_wmma_f32_16x16x32_bf16(false, pa, false, vb,
                                                      (short)0, o[nt], false, false);
    }
  }

  // finalize: divide by row sums, write bf16 context at permuted columns
#pragma unroll
  for (int r = 0; r < 8; ++r) {
    float inv = 1.0f / lrow[r];
    int qi = m0 + rbase + r;
    bf16_t* orow = O + (size_t)qi * (NHEADS * HDIM) + h * HDIM;
#pragma unroll
    for (int nt = 0; nt < 8; ++nt)
      orow[pcol(nt * 16 + nl)] = f2bf(o[nt][r] * inv);
  }
}

extern "C" void kernel_launch(void* const* d_in, const int* in_sizes, int n_in,
                              void* d_out, int out_size, void* d_ws, size_t ws_size,
                              hipStream_t stream) {
  const int*   positions = (const int*)d_in[0];
  const float* hs = (const float*)d_in[1];
  const float* wq = (const float*)d_in[2];
  const float* wk = (const float*)d_in[3];
  const float* wv = (const float*)d_in[4];
  const float* wo = (const float*)d_in[5];
  float* out = (float*)d_out;

  char* ws = (char*)d_ws;
  size_t off = 0;
  auto alloc = [&](size_t elems) { bf16_t* p = (bf16_t*)(ws + off); off += elems * 2; return p; };
  bf16_t* hsb = alloc((size_t)SEQ * HIDDEN);        // hidden bf16, perm-K
  bf16_t* wqT = alloc((size_t)4096 * HIDDEN);       // [N][K-perm]
  bf16_t* wkT = alloc((size_t)1024 * HIDDEN);
  bf16_t* wvT = alloc((size_t)1024 * HIDDEN);
  bf16_t* woT = alloc((size_t)HIDDEN * 4096);
  bf16_t* qb  = alloc((size_t)SEQ * NHEADS * HDIM); // perm-d
  bf16_t* kb  = alloc((size_t)SEQ * NKV * HDIM);    // perm-d
  bf16_t* vb  = alloc((size_t)SEQ * NKV * HDIM);    // natural
  bf16_t* vT  = alloc((size_t)NKV * HDIM * SEQ);    // perm-t
  bf16_t* ab  = alloc((size_t)SEQ * NHEADS * HDIM); // perm-col (K for final GEMM)

  dim3 tb(32, 8);
  cvt_bf16_kernel<<<(SEQ * HIDDEN + 255) / 256, 256, 0, stream>>>(hs, hsb, SEQ * HIDDEN);
  transpose_cvt_kernel<<<dim3(4096 / 32, HIDDEN / 32), tb, 0, stream>>>(wq, wqT, HIDDEN, 4096);
  transpose_cvt_kernel<<<dim3(1024 / 32, HIDDEN / 32), tb, 0, stream>>>(wk, wkT, HIDDEN, 1024);
  transpose_cvt_kernel<<<dim3(1024 / 32, HIDDEN / 32), tb, 0, stream>>>(wv, wvT, HIDDEN, 1024);
  transpose_cvt_kernel<<<dim3(4096 / 32, 4096 / 32), tb, 0, stream>>>(wo, woT, 4096, 4096);

  gemm_bf16_kernel<true,  true ><<<dim3(4096 / 128, SEQ / 128), 256, 0, stream>>>(hsb, wqT, qb, SEQ, 4096, HIDDEN);
  gemm_bf16_kernel<true,  true ><<<dim3(1024 / 128, SEQ / 128), 256, 0, stream>>>(hsb, wkT, kb, SEQ, 1024, HIDDEN);
  gemm_bf16_kernel<true,  false><<<dim3(1024 / 128, SEQ / 128), 256, 0, stream>>>(hsb, wvT, vb, SEQ, 1024, HIDDEN);

  rope_kernel<<<(SEQ * NHEADS * 64 + 255) / 256, 256, 0, stream>>>(positions, qb, NHEADS);
  rope_kernel<<<(SEQ * NKV * 64 + 255) / 256, 256, 0, stream>>>(positions, kb, NKV);

  vtrans_kernel<<<(NKV * HDIM * SEQ + 255) / 256, 256, 0, stream>>>(vb, vT);

  attn_kernel<<<dim3(SEQ / 128, NHEADS), 256, 0, stream>>>(qb, kb, vT, ab);

  gemm_bf16_kernel<false, false><<<dim3(4096 / 128, SEQ / 128), 256, 0, stream>>>(ab, woT, out, SEQ, 4096, 4096);
}
